// RetNetAttention_88390426952203
// MI455X (gfx1250) — matrix operand order repaired and checked by reading, not compiled, not run on hardware
//
#include <hip/hip_runtime.h>
#include <hip/hip_bf16.h>
#include <math.h>

// Problem constants (match reference)
#define B_      4
#define L_      2048
#define D_      1024
#define H_      8
#define DH_     128
#define CHUNK_  64
#define ESPLIT_ 4
#define EW_     (DH_ / ESPLIT_)   // 32 state columns per block
#define EPS_    1e-5f

typedef __bf16 bf16;
typedef bf16  v16bf __attribute__((ext_vector_type(16)));
typedef bf16  v8bf  __attribute__((ext_vector_type(8)));
typedef float v8f   __attribute__((ext_vector_type(8)));

// ---------------------------------------------------------------------------
// CDNA5 WMMA fragment loaders (wave32, 16x16x32 bf16).
// A (16xK): lanes 0-15 row M=lane hold K k0+0..7 (v0..3) and k0+16..23 (v4..7),
//           lanes 16-31 same rows hold K k0+8..15 and k0+24..31.
// B (Kx16): same K split as A: lanes 0-15 col N=lane hold K k0+0..7, k0+16..23; lanes 16-31 hold k0+8..15, k0+24..31.
// ---------------------------------------------------------------------------
static __device__ __forceinline__ v16bf load_a_frag(const bf16* base, int ld,
                                                    int row0, int k0, int lane) {
  const int hl = lane >> 4, r = lane & 15;
  const bf16* p = base + (size_t)(row0 + r) * ld + k0 + hl * 8;
  v16bf f;
  ((v8bf*)&f)[0] = *(const v8bf*)(p);
  ((v8bf*)&f)[1] = *(const v8bf*)(p + 16);
  return f;
}

static __device__ __forceinline__ v16bf load_b_frag(const bf16* base, int ld,
                                                    int col0, int k0, int lane) {
  const int hl = lane >> 4, r = lane & 15;
  const bf16* p = base + (size_t)(col0 + r) * ld + k0 + hl * 8;
  v16bf f;
  ((v8bf*)&f)[0] = *(const v8bf*)(p);
  ((v8bf*)&f)[1] = *(const v8bf*)(p + 16);
  return f;
}

static __device__ __forceinline__ v8f wmma_bf16(v16bf a, v16bf b, v8f c) {
  return __builtin_amdgcn_wmma_f32_16x16x32_bf16(false, a, false, b,
                                                 (short)0, c, false, false);
}

// ---------------------------------------------------------------------------
// f32 -> bf16 cast (grid-stride)
// ---------------------------------------------------------------------------
__global__ __launch_bounds__(256) void cast_kernel(const float* __restrict__ src,
                                                   bf16* __restrict__ dst, size_t n) {
  for (size_t i = (size_t)blockIdx.x * blockDim.x + threadIdx.x; i < n;
       i += (size_t)gridDim.x * blockDim.x)
    dst[i] = (bf16)src[i];
}

// ---------------------------------------------------------------------------
// GEMM: C[M,N] = A[M,K] @ W[N,K]^T.  Block = 8 waves, tile 128(M) x 128(N);
// each wave owns a 16x128 strip (one A-frag reused across 8 B-frags).
// MODE 0: write bf16 remapped to [b][h][l][e]  (QKV projections, z = 0..2)
// MODE 1: write f32 row-major                 (final Wo projection)
// ---------------------------------------------------------------------------
template <int MODE>
__global__ __launch_bounds__(256) void gemm_kernel(const bf16* __restrict__ A,
                                                   const bf16* __restrict__ W,
                                                   void* __restrict__ out,
                                                   int M, int N, int K) {
  const int lane = threadIdx.x & 31;
  const int wave = threadIdx.x >> 5;
  const int m0 = blockIdx.x * 128 + wave * 16;
  const int n0 = blockIdx.y * 128;
  const bf16* Wz = W + (size_t)blockIdx.z * N * K;

  v8f acc[8] = {};
  for (int k0 = 0; k0 < K; k0 += 32) {
    v16bf a = load_a_frag(A, K, m0, k0, lane);
#pragma unroll
    for (int t = 0; t < 8; ++t) {
      v16bf b = load_b_frag(Wz, K, n0 + t * 16, k0, lane);
      acc[t] = wmma_bf16(a, b, acc[t]);
    }
  }

  const int hl = lane >> 4, r = lane & 15;
#pragma unroll
  for (int t = 0; t < 8; ++t) {
#pragma unroll
    for (int i = 0; i < 8; ++i) {
      const int m = m0 + i + 8 * hl;       // C/D layout: vgpr i -> M = i (+8 hi half)
      const int n = n0 + t * 16 + r;       // N = lane & 15
      const float v = acc[t][i];
      if (MODE == 0) {
        const int b = m / L_, l = m % L_;
        const int h = n / DH_, e = n % DH_;
        bf16* dst = (bf16*)out + (size_t)blockIdx.z * M * N;
        dst[(((size_t)b * H_ + h) * L_ + l) * DH_ + e] = (bf16)v;
      } else {
        ((float*)out)[(size_t)m * N + n] = v;
      }
    }
  }
}

// ---------------------------------------------------------------------------
// Chunkwise-parallel retention.  One block per (b, h, e-slice of 32 cols).
// State master S32 (f32) + WMMA operand copy SbfT (bf16, transposed) in LDS.
// Per chunk: P = mask(Q K^T), Y = gamma^{i+1} Q S + P V, S = gamma^C S + K'^T V
// Decay powers gamma^n (n=0..64) come from an LDS LUT (no transcendentals in
// the chunk loop).  Group-norm partials written per (bh, e-slice): no atomics.
// ---------------------------------------------------------------------------
__global__ __launch_bounds__(256) void retnet_kernel(const bf16* __restrict__ q_g,
                                                     const bf16* __restrict__ k_g,
                                                     const bf16* __restrict__ v_g,
                                                     float* __restrict__ Y,
                                                     float* __restrict__ stats) {
  __shared__ float S32[DH_ * EW_];        // [d][e]  16 KB  (f32 state master)
  __shared__ bf16  SbfT[EW_ * DH_];       // [e][d]   8 KB  (B-operand copy)
  __shared__ bf16  KtT[DH_ * CHUNK_];     // [d][j]  16 KB  (decay-scaled K^T)
  __shared__ bf16  Vt[EW_ * CHUNK_];      // [e][j]   4 KB  (V^T slice)
  __shared__ bf16  Ps[CHUNK_ * CHUNK_];   // [i][j]   8 KB  (masked scores)
  __shared__ float gpow[CHUNK_ + 1];      // gamma^n LUT
  __shared__ float red[512];

  const int tid = threadIdx.x;
  const int lane = tid & 31;
  const int wave = tid >> 5;
  const int bh = blockIdx.x / ESPLIT_;
  const int es = blockIdx.x % ESPLIT_;
  const int e0 = es * EW_;
  const int h = bh % H_;
  const int hl = lane >> 4, r = lane & 15;

  const float lg0 = logf(1.0f / 32.0f);
  const float lg1 = logf(1.0f / 512.0f);
  const float gamma = 1.0f - expf(lg0 + (lg1 - lg0) * ((float)h / (float)(H_ - 1)));

  const bf16* q_bh = q_g + (size_t)bh * L_ * DH_;
  const bf16* k_bh = k_g + (size_t)bh * L_ * DH_;
  const bf16* v_bh = v_g + (size_t)bh * L_ * DH_;
  float* Y_bh = Y + (size_t)bh * L_ * DH_;

  for (int i = tid; i < DH_ * EW_; i += 256) S32[i] = 0.0f;
  for (int i = tid; i < EW_ * DH_; i += 256) SbfT[i] = (bf16)0.0f;
  for (int i = tid; i <= CHUNK_; i += 256) gpow[i] = __powf(gamma, (float)i);
  __syncthreads();

  const float gC = gpow[CHUNK_];
  float lsum = 0.0f, lsq = 0.0f;

  for (int c = 0; c < L_ / CHUNK_; ++c) {
    const bf16* kc = k_bh + (size_t)c * CHUNK_ * DH_;
    const bf16* vc = v_bh + (size_t)c * CHUNK_ * DH_;

    // Stage K'^T (scaled by gamma^{C-1-j}) and V^T slice into LDS.
    for (int idx = tid; idx < DH_ * CHUNK_; idx += 256) {
      const int j = idx / DH_, d = idx % DH_;
      const float kv = (float)kc[idx] * gpow[CHUNK_ - 1 - j];
      KtT[d * CHUNK_ + j] = (bf16)kv;
    }
    for (int idx = tid; idx < EW_ * CHUNK_; idx += 256) {
      const int j = idx / EW_, e = idx % EW_;
      Vt[e * CHUNK_ + j] = vc[j * DH_ + e0 + e];
    }
    __syncthreads();

    // Phase B: scores P = (Q K^T) * decay mask.  16 tiles, 2 per wave.
    for (int tt = 0; tt < 2; ++tt) {
      const int t = wave * 2 + tt;
      const int it = t >> 2, jt = t & 3;
      v8f acc = {};
#pragma unroll
      for (int kk = 0; kk < DH_ / 32; ++kk) {
        v16bf a = load_a_frag(q_bh, DH_, c * CHUNK_ + it * 16, kk * 32, lane);
        v16bf b = load_b_frag(kc, DH_, jt * 16, kk * 32, lane);
        acc = wmma_bf16(a, b, acc);
      }
#pragma unroll
      for (int i8 = 0; i8 < 8; ++i8) {
        const int i = it * 16 + i8 + 8 * hl;
        const int j = jt * 16 + r;
        const float v = (i >= j) ? acc[i8] * gpow[i - j] : 0.0f;
        Ps[i * CHUNK_ + j] = (bf16)v;
      }
    }
    __syncthreads();

    // Phase C: Y = gamma^{i+1} * (Q @ S_prev) + P @ V.  One tile per wave.
    {
      const int it = wave >> 1, et = wave & 1;
      v8f acc = {};
#pragma unroll
      for (int kk = 0; kk < DH_ / 32; ++kk) {
        v16bf a = load_a_frag(q_bh, DH_, c * CHUNK_ + it * 16, kk * 32, lane);
        v16bf b = load_b_frag(SbfT, DH_, et * 16, kk * 32, lane);
        acc = wmma_bf16(a, b, acc);
      }
#pragma unroll
      for (int i8 = 0; i8 < 8; ++i8) {
        const int i = it * 16 + i8 + 8 * hl;
        acc[i8] *= gpow[i + 1];
      }
#pragma unroll
      for (int kk = 0; kk < CHUNK_ / 32; ++kk) {
        v16bf a = load_a_frag(Ps, CHUNK_, it * 16, kk * 32, lane);
        v16bf b = load_b_frag(Vt, CHUNK_, et * 16, kk * 32, lane);
        acc = wmma_bf16(a, b, acc);
      }
#pragma unroll
      for (int i8 = 0; i8 < 8; ++i8) {
        const int i = it * 16 + i8 + 8 * hl;
        const int e = et * 16 + r;
        const float y = acc[i8];
        Y_bh[(size_t)(c * CHUNK_ + i) * DH_ + e0 + e] = y;
        lsum += y;
        lsq += y * y;
      }
    }
    __syncthreads();

    // Phase D: S = gamma^C * S + K'^T @ V.  Wave owns d-rows [16w, 16w+16).
    {
      const int d0 = wave * 16;
      for (int et = 0; et < 2; ++et) {
        v8f acc = {};
#pragma unroll
        for (int kk = 0; kk < CHUNK_ / 32; ++kk) {
          v16bf a = load_a_frag(KtT, CHUNK_, d0, kk * 32, lane);
          v16bf b = load_b_frag(Vt, CHUNK_, et * 16, kk * 32, lane);
          acc = wmma_bf16(a, b, acc);
        }
#pragma unroll
        for (int i8 = 0; i8 < 8; ++i8) {
          const int d = d0 + i8 + 8 * hl;
          const int e = et * 16 + r;
          const float s = gC * S32[d * EW_ + e] + acc[i8];
          S32[d * EW_ + e] = s;
          SbfT[e * DH_ + d] = (bf16)s;
        }
      }
    }
    __syncthreads();
  }

  // Block reduction of group-norm partials; one slot per (bh, e-slice).
  red[tid] = lsum;
  red[256 + tid] = lsq;
  __syncthreads();
  for (int s = 128; s > 0; s >>= 1) {
    if (tid < s) {
      red[tid] += red[tid + s];
      red[256 + tid] += red[256 + tid + s];
    }
    __syncthreads();
  }
  if (tid == 0) {
    stats[(bh * ESPLIT_ + es) * 2 + 0] = red[0];
    stats[(bh * ESPLIT_ + es) * 2 + 1] = red[256];
  }
}

// ---------------------------------------------------------------------------
// Group-norm over (L, dh) per (b, h) + affine; emits bf16 rows for Wo GEMM.
// Sums the 4 e-slice partials in fixed order (deterministic).
// ---------------------------------------------------------------------------
__global__ __launch_bounds__(256) void norm_kernel(const float* __restrict__ Y,
                                                   const float* __restrict__ stats,
                                                   const float* __restrict__ gnw,
                                                   const float* __restrict__ gnb,
                                                   bf16* __restrict__ X2) {
  const float cnt = (float)L_ * (float)DH_;
  for (size_t idx = (size_t)blockIdx.x * blockDim.x + threadIdx.x;
       idx < (size_t)B_ * L_ * D_; idx += (size_t)gridDim.x * blockDim.x) {
    const int col = (int)(idx % D_);
    const size_t row = idx / D_;
    const int l = (int)(row % L_);
    const int b = (int)(row / L_);
    const int hh = col / DH_, e = col % DH_;
    const int bh = b * H_ + hh;
    float s0 = 0.0f, s1 = 0.0f;
#pragma unroll
    for (int es = 0; es < ESPLIT_; ++es) {
      s0 += stats[(bh * ESPLIT_ + es) * 2 + 0];
      s1 += stats[(bh * ESPLIT_ + es) * 2 + 1];
    }
    const float mean = s0 / cnt;
    const float var = s1 / cnt - mean * mean;
    const float rstd = rsqrtf(var + EPS_);
    const float y = Y[((size_t)bh * L_ + l) * DH_ + e];
    X2[idx] = (bf16)((y - mean) * rstd * gnw[col] + gnb[col]);
  }
}

// ---------------------------------------------------------------------------
// Host-side orchestration
// ---------------------------------------------------------------------------
extern "C" void kernel_launch(void* const* d_in, const int* in_sizes, int n_in,
                              void* d_out, int out_size, void* d_ws, size_t ws_size,
                              hipStream_t stream) {
  (void)in_sizes; (void)n_in; (void)out_size; (void)ws_size;
  const float* x   = (const float*)d_in[0];
  const float* Wq  = (const float*)d_in[1];
  const float* Wk  = (const float*)d_in[2];
  const float* Wv  = (const float*)d_in[3];
  const float* Wo  = (const float*)d_in[4];
  const float* gnw = (const float*)d_in[5];
  const float* gnb = (const float*)d_in[6];
  float* out = (float*)d_out;

  const size_t BLD = (size_t)B_ * L_ * D_;   // 8,388,608
  const size_t DD  = (size_t)D_ * D_;        // 1,048,576
  const int    M   = B_ * L_;                // 8192

  char* ws = (char*)d_ws;
  size_t off = 0;
  auto aln = [](size_t v) { return (v + 255) & ~(size_t)255; };
  bf16*  Xbf   = (bf16*)(ws + off);  off = aln(off + BLD * 2);
  bf16*  Wbf   = (bf16*)(ws + off);  off = aln(off + 4 * DD * 2);
  bf16*  QKV   = (bf16*)(ws + off);  off = aln(off + 3 * BLD * 2);
  float* Yws   = (float*)(ws + off); off = aln(off + BLD * 4);
  bf16*  X2    = (bf16*)(ws + off);  off = aln(off + BLD * 2);
  float* stats = (float*)(ws + off);

  // 1) cast activations + weights to bf16
  cast_kernel<<<1024, 256, 0, stream>>>(x, Xbf, BLD);
  cast_kernel<<<256, 256, 0, stream>>>(Wq, Wbf + 0 * DD, DD);
  cast_kernel<<<256, 256, 0, stream>>>(Wk, Wbf + 1 * DD, DD);
  cast_kernel<<<256, 256, 0, stream>>>(Wv, Wbf + 2 * DD, DD);
  cast_kernel<<<256, 256, 0, stream>>>(Wo, Wbf + 3 * DD, DD);

  // 2) Q/K/V projections (bf16 WMMA GEMM, epilogue remaps to [b][h][l][e])
  gemm_kernel<0><<<dim3(M / 128, D_ / 128, 3), 256, 0, stream>>>(Xbf, Wbf, QKV,
                                                                 M, D_, D_);

  // 3) chunkwise retention + group-norm partial stats (no atomics)
  retnet_kernel<<<B_ * H_ * ESPLIT_, 256, 0, stream>>>(QKV, QKV + BLD,
                                                       QKV + 2 * BLD, Yws, stats);

  // 4) group-norm + affine, emit bf16
  norm_kernel<<<2048, 256, 0, stream>>>(Yws, stats, gnw, gnb, X2);

  // 5) output projection out = X2 @ Wo^T (f32 out)
  gemm_kernel<1><<<dim3(M / 128, D_ / 128, 1), 256, 0, stream>>>(X2, Wbf + 3 * DD,
                                                                 out, M, D_, D_);
}
